// DeepInterestNetwork_57604101374603
// MI455X (gfx1250) — matrix operand order repaired and checked by reading, not compiled, not run on hardware
//
#include <hip/hip_runtime.h>
#include <hip/hip_bf16.h>
#include <math.h>

typedef _Float16 v16h __attribute__((ext_vector_type(16)));
typedef _Float16 v8h  __attribute__((ext_vector_type(8)));
typedef float    v8f  __attribute__((ext_vector_type(8)));

#define T_SEQ 200
#define T_PAD 208
#define EMB_E 64
#define KSTR  72    // 64 + 8 halfs pad: 144B rows, 16B aligned, conflict-free b128 reads
#define BSTR  136   // 128 + 8 halfs pad: 272B rows

__device__ __forceinline__ v16h cat8(v8h lo, v8h hi) {
  return __builtin_shufflevector(lo, hi, 0,1,2,3,4,5,6,7,8,9,10,11,12,13,14,15);
}

__global__ __launch_bounds__(256) void din_kernel(
    const int* __restrict__ query, const int* __restrict__ keys,
    const float* __restrict__ emb,
    const float* __restrict__ att_w1, const float* __restrict__ att_b1,
    const float* __restrict__ att_w2, const float* __restrict__ att_b2,
    const float* __restrict__ deep_w1, const float* __restrict__ deep_b1,
    const float* __restrict__ deep_w2, const float* __restrict__ deep_b2,
    const float* __restrict__ out_w, const float* __restrict__ out_b,
    float* __restrict__ out)
{
  __shared__ __align__(16) _Float16 kLds[T_PAD * KSTR];    // k rows, f16
  __shared__ __align__(16) _Float16 Bcol[EMB_E * BSTR];    // [n][0:64]=W1-W2, [n][64:128]=W3 (col-major)
  __shared__ __align__(16) _Float16 qh[EMB_E];
  __shared__ float scorePart[4][T_PAD];
  __shared__ float wLds[T_PAD];
  __shared__ float qf[EMB_E], bias2[EMB_E], w2f[EMB_E];
  __shared__ float iPart[4][EMB_E], interest[EMB_E];
  __shared__ float d1v[128], d2v[64];
  __shared__ float red[8], blkred[2];

  const int b   = blockIdx.x;
  const int tid = threadIdx.x;

  // ---- stage q, att_w2 ----
  if (tid < 64) {
    float qv = emb[(long)query[b] * EMB_E + tid];
    qf[tid] = qv;
    qh[tid] = (_Float16)qv;
    w2f[tid] = att_w2[tid];
  }
  __syncthreads();

  // ---- fold q@(W0+W2) into bias ----
  if (tid < 64) {
    float acc = att_b1[tid];
    for (int i = 0; i < 64; ++i)
      acc += qf[i] * (att_w1[i * 64 + tid] + att_w1[(128 + i) * 64 + tid]);
    bias2[tid] = acc;
  }

  // ---- build B col-major: rows 0..63 = W1-W2, 64..127 = W3 ----
  for (int idx = tid; idx < 64 * 128; idx += 256) {
    int n = idx >> 7, kk = idx & 127;
    float v = (kk < 64)
      ? (att_w1[(64 + kk) * 64 + n] - att_w1[(128 + kk) * 64 + n])
      : att_w1[(192 + (kk - 64)) * 64 + n];
    Bcol[n * BSTR + kk] = (_Float16)v;
  }

  // ---- gather k rows -> f16 LDS (coalesced 256B per row) ----
  {
    int e = tid & 63, tg = tid >> 6;
    for (int t = tg; t < T_SEQ; t += 4)
      kLds[t * KSTR + e] = (_Float16)emb[(long)keys[b * T_SEQ + t] * EMB_E + e];
    for (int t = T_SEQ + tg; t < T_PAD; t += 4)
      kLds[t * KSTR + e] = (_Float16)0.0f;
  }
  __syncthreads();

  // ---- WMMA:  h = relu(k@(W1-W2) + (q*k)@W3 + bias2);  scorePart = h @ w2 ----
  {
    const int wid   = tid >> 5;
    const int lane  = tid & 31;
    const int ntile = wid & 3;       // 4 N tiles of 16
    const int mhalf = wid >> 2;      // 2-way split of 13 M tiles
    const int nloc  = lane & 15;
    const int n     = ntile * 16 + nloc;
    const int kb    = (lane >> 4) ? 8 : 0;    // A-frag K sub-offset (halfs)
    const int kbB   = (lane >> 4) ? 8 : 0;    // B-frag K sub-offset (halfs)

    // B fragments (fixed per wave): two k-steps x {Wk, Wm}
    const _Float16* bp = &Bcol[n * BSTR];
    v16h bk0 = cat8(*(const v8h*)(bp +  0 + kbB), *(const v8h*)(bp + 16 + kbB));
    v16h bk1 = cat8(*(const v8h*)(bp + 32 + kbB), *(const v8h*)(bp + 48 + kbB));
    v16h bm0 = cat8(*(const v8h*)(bp + 64 + kbB), *(const v8h*)(bp + 80 + kbB));
    v16h bm1 = cat8(*(const v8h*)(bp + 96 + kbB), *(const v8h*)(bp + 112 + kbB));
    // q fragments (A layout, loop invariant)
    v16h q0 = cat8(*(const v8h*)(qh + 0  + kb), *(const v8h*)(qh + 16 + kb));
    v16h q1 = cat8(*(const v8h*)(qh + 32 + kb), *(const v8h*)(qh + 48 + kb));

    const float bn  = bias2[n];
    const float w2n = w2f[n];

    for (int mt = mhalf; mt < 13; mt += 2) {
      const _Float16* rp = &kLds[(mt * 16 + nloc) * KSTR];
      v16h a0 = cat8(*(const v8h*)(rp + 0  + kb), *(const v8h*)(rp + 16 + kb));
      v16h a1 = cat8(*(const v8h*)(rp + 32 + kb), *(const v8h*)(rp + 48 + kb));
      v16h a0q = a0 * q0;   // (q .* k) fragment, packed f16 mul
      v16h a1q = a1 * q1;

      v8f c = {};
      c = __builtin_amdgcn_wmma_f32_16x16x32_f16(false, a0,  false, bk0, (short)0, c, false, false);
      c = __builtin_amdgcn_wmma_f32_16x16x32_f16(false, a0q, false, bm0, (short)0, c, false, false);
      c = __builtin_amdgcn_wmma_f32_16x16x32_f16(false, a1,  false, bk1, (short)0, c, false, false);
      c = __builtin_amdgcn_wmma_f32_16x16x32_f16(false, a1q, false, bm1, (short)0, c, false, false);

      // fused bias + relu + (h @ att_w2): reduce over the 16 N-lanes
      float p[8];
      #pragma unroll
      for (int j = 0; j < 8; ++j) {
        float h = c[j] + bn;
        h = h > 0.0f ? h : 0.0f;
        p[j] = h * w2n;
      }
      #pragma unroll
      for (int j = 0; j < 8; ++j) {
        float v = p[j];
        v += __shfl_xor(v, 1, 32);
        v += __shfl_xor(v, 2, 32);
        v += __shfl_xor(v, 4, 32);
        v += __shfl_xor(v, 8, 32);
        p[j] = v;
      }
      if (nloc == 0) {  // lanes 0 (rows m..m+7) and 16 (rows m+8..m+15)
        int mbase = mt * 16 + ((lane >> 4) ? 8 : 0);
        #pragma unroll
        for (int j = 0; j < 8; ++j) scorePart[ntile][mbase + j] = p[j];
      }
    }
  }
  __syncthreads();

  // ---- softmax over T ----
  {
    float sv = -1e30f;
    if (tid < T_SEQ)
      sv = scorePart[0][tid] + scorePart[1][tid] + scorePart[2][tid]
         + scorePart[3][tid] + att_b2[0];
    float m = sv;
    for (int o = 16; o >= 1; o >>= 1) m = fmaxf(m, __shfl_xor(m, o, 32));
    if ((tid & 31) == 0) red[tid >> 5] = m;
    __syncthreads();
    if (tid == 0) {
      float mm = red[0];
      for (int i = 1; i < 8; ++i) mm = fmaxf(mm, red[i]);
      blkred[0] = mm;
    }
    __syncthreads();
    float ev = (tid < T_SEQ) ? __expf(sv - blkred[0]) : 0.0f;
    float s = ev;
    for (int o = 16; o >= 1; o >>= 1) s += __shfl_xor(s, o, 32);
    if ((tid & 31) == 0) red[tid >> 5] = s;
    __syncthreads();
    if (tid == 0) {
      float ss = 0.0f;
      for (int i = 0; i < 8; ++i) ss += red[i];
      blkred[1] = ss;
    }
    __syncthreads();
    if (tid < T_PAD) wLds[tid] = ev / blkred[1];
  }
  __syncthreads();

  // ---- interest = sum_t w[t] * k[t] (reuse f16 k in LDS) ----
  {
    int e = tid & 63, g = tid >> 6;
    float acc = 0.0f;
    for (int t = g * 50; t < (g + 1) * 50; ++t)
      acc += wLds[t] * (float)kLds[t * KSTR + e];
    iPart[g][e] = acc;
  }
  __syncthreads();
  if (tid < 64)
    interest[tid] = iPart[0][tid] + iPart[1][tid] + iPart[2][tid] + iPart[3][tid];
  __syncthreads();

  // ---- deep head ----
  if (tid < 128) {
    float acc = deep_b1[tid];
    for (int i = 0; i < 64; ++i) acc += interest[i] * deep_w1[i * 128 + tid];
    d1v[tid] = acc > 0.0f ? acc : 0.0f;
  }
  __syncthreads();
  if (tid < 64) {
    float acc = deep_b2[tid];
    for (int i = 0; i < 128; ++i) acc += d1v[i] * deep_w2[i * 64 + tid];
    d2v[tid] = acc > 0.0f ? acc : 0.0f;
  }
  __syncthreads();
  if (tid == 0) {
    float acc = out_b[0];
    for (int j = 0; j < 64; ++j) acc += d2v[j] * out_w[j];
    out[b] = 1.0f / (1.0f + __expf(-acc));
  }
}

extern "C" void kernel_launch(void* const* d_in, const int* in_sizes, int n_in,
                              void* d_out, int out_size, void* d_ws, size_t ws_size,
                              hipStream_t stream) {
  (void)n_in; (void)d_ws; (void)ws_size; (void)out_size;
  const int*   query   = (const int*)d_in[0];
  const int*   keys    = (const int*)d_in[1];
  const float* emb     = (const float*)d_in[2];
  const float* att_w1  = (const float*)d_in[3];
  const float* att_b1  = (const float*)d_in[4];
  const float* att_w2  = (const float*)d_in[5];
  const float* att_b2  = (const float*)d_in[6];
  const float* deep_w1 = (const float*)d_in[7];
  const float* deep_b1 = (const float*)d_in[8];
  const float* deep_w2 = (const float*)d_in[9];
  const float* deep_b2 = (const float*)d_in[10];
  const float* out_w   = (const float*)d_in[11];
  const float* out_b   = (const float*)d_in[12];
  float* out = (float*)d_out;

  const int B = in_sizes[0];  // 4096
  din_kernel<<<B, 256, 0, stream>>>(query, keys, emb,
                                    att_w1, att_b1, att_w2, att_b2,
                                    deep_w1, deep_b1, deep_w2, deep_b2,
                                    out_w, out_b, out);
}
